// PaiNN_9414568313173
// MI455X (gfx1250) — matrix operand order repaired and checked by reading, not compiled, not run on hardware
//
#include <hip/hip_runtime.h>
#include <hip/hip_bf16.h>

typedef __attribute__((ext_vector_type(16))) _Float16 v16h;
typedef __attribute__((ext_vector_type(8)))  float    v8f;
typedef __attribute__((ext_vector_type(8)))  unsigned int v8u;

#define N_ELEC 1024
#define N_NUC  256
#define DD     64
#define FF     64
#define EDG    131072
#define HHH    128
#define HGG    128
#define TD     192   // 3*D

#ifndef __has_builtin
#define __has_builtin(x) 0
#endif
#if defined(__AMDGCN__) && __has_builtin(__builtin_amdgcn_tensor_load_to_lds)
#define USE_TDM 1
#else
#define USE_TDM 0
#endif

// Staged dist chunk: 64 edges x 64 f32, TDM-padded 2 dwords per 64 -> row stride 66 dw
#define CH_EDGES   64
#define ROW_DW     66
#define STAGE_DW   (CH_EDGES * ROW_DW)        // 4224 dwords per buffer
#define CHUNKS     8                          // 8 x 64 = 512 edges per block

#if USE_TDM
typedef unsigned int u32x4 __attribute__((ext_vector_type(4)));
typedef int          i32x8 __attribute__((ext_vector_type(8)));
typedef int          i32x4 __attribute__((ext_vector_type(4)));

// 1-D TDM copy of 4096 f32 (16 KB) global -> LDS, padding 2 dwords every 64.
__device__ __forceinline__ void tdm_load_chunk(unsigned lds_byte_off, const float* gsrc) {
    unsigned long long ga = (unsigned long long)gsrc;
    u32x4 g0;
    g0[0] = 1u;                                          // count=1 (valid user D#)
    g0[1] = lds_byte_off;                                // lds_addr (bytes)
    g0[2] = (unsigned)(ga & 0xffffffffu);                // global_addr[31:0]
    g0[3] = (unsigned)((ga >> 32) & 0x01ffffffu)         // global_addr[56:32]
          | (2u << 30);                                  // type = 2 ("image")
    i32x8 g1;
    g1[0] = (int)(0x00020000u       // data_size = 2 (4 bytes)
          | (1u << 20)              // pad_enable
          | (5u << 22)              // pad_interval: 64 dwords
          | (1u << 25));            // pad_amount: 2 dwords
    g1[1] = (int)(4096u << 16);     // tensor_dim0[15:0] << 16 (dim0 = 4096)
    g1[2] = (int)(1u << 16);        // tensor_dim0 hi = 0; tensor_dim1 = 1
    g1[3] = (int)(4096u << 16);     // tensor_dim1 hi = 0; tile_dim0 = 4096
    g1[4] = 0;                      // tile_dim1 = 0, tile_dim2 = 0
    g1[5] = 4096;                   // tensor_dim0_stride low
    g1[6] = 0;
    g1[7] = 0;
    i32x4 g2 = {0, 0, 0, 0};
    i32x4 g3 = {0, 0, 0, 0};
#if __clang_major__ >= 23
    i32x8 g4 = {0, 0, 0, 0, 0, 0, 0, 0};
    __builtin_amdgcn_tensor_load_to_lds(g0, g1, g2, g3, g4, 0);
#else
    __builtin_amdgcn_tensor_load_to_lds(g0, g1, g2, g3, 0);
#endif
}
#endif  // USE_TDM

// ---------------------------------------------------------------------------
// Kernel 1: zero accumulators, seed output with residuals
// ---------------------------------------------------------------------------
__global__ void __launch_bounds__(256) painn_init(const float* __restrict__ s_elec,
                                                  const float* __restrict__ v_elec,
                                                  float* __restrict__ out,
                                                  float* __restrict__ zs,
                                                  float* __restrict__ zv) {
    int i = blockIdx.x * 256 + threadIdx.x;
    if (i < N_ELEC * 256) {
        int n = i >> 8, c = i & 255;
        out[i] = (c < DD) ? s_elec[n * DD + c] : v_elec[n * TD + (c - DD)];
    }
    if (i < 3 * N_ELEC * DD) zs[i] = 0.0f;
    if (i < 3 * N_ELEC * TD) zv[i] = 0.0f;
}

// ---------------------------------------------------------------------------
// Kernel 2: h_all[t][node][0..191] = silu(s @ W_h1[t]) @ W_h2[t]
// ---------------------------------------------------------------------------
__global__ void __launch_bounds__(192) painn_h(const float* __restrict__ s_elec,
                                               const float* __restrict__ s_nuc,
                                               const float* __restrict__ W_h1,
                                               const float* __restrict__ W_h2,
                                               float* __restrict__ h_all) {
    int b = blockIdx.x;
    int t, node; const float* s;
    if (b < N_NUC)                { t = 1; node = b;                   s = s_nuc  + node * DD; }
    else if (b < N_NUC + N_ELEC)  { t = 2; node = b - N_NUC;           s = s_elec + node * DD; }
    else                          { t = 3; node = b - N_NUC - N_ELEC;  s = s_elec + node * DD; }
    const float* w1 = W_h1 + (size_t)t * DD * HHH;
    const float* w2 = W_h2 + (size_t)t * HHH * TD;
    __shared__ float s_l[DD];
    __shared__ float tmp[HHH];
    int tid = threadIdx.x;               // 192 threads
    if (tid < DD) s_l[tid] = s[tid];
    __syncthreads();
    if (tid < HHH) {
        float a = 0.0f;
        for (int d = 0; d < DD; ++d) a += s_l[d] * w1[d * HHH + tid];
        tmp[tid] = a / (1.0f + __expf(-a));     // silu
    }
    __syncthreads();
    float a = 0.0f;
    for (int h = 0; h < HHH; ++h) a += tmp[h] * w2[h * TD + tid];
    h_all[((size_t)(t - 1) * N_ELEC + node) * TD + tid] = a;
}

// ---------------------------------------------------------------------------
// Kernel 3: edge GEMM (WMMA) with TDM-staged dist + scatter-add segment sums
//   grid: 3 types x 256 blocks; block = 128 thr = 4 waves
//   block = 512 edges, in 8 double-buffered chunks of 64 edges
// ---------------------------------------------------------------------------
__global__ void __launch_bounds__(128) painn_edge(const float* __restrict__ dist,
                                                  const float* __restrict__ dirs,
                                                  const float* __restrict__ W_w,
                                                  const float* __restrict__ v_elec,
                                                  const float* __restrict__ v_nuc,
                                                  const int*   __restrict__ senders,
                                                  const int*   __restrict__ receivers,
                                                  const float* __restrict__ h_all,
                                                  float* __restrict__ zs,
                                                  float* __restrict__ zv) {
    // B operand (W_w^f16) pre-packed in exact wmma B-register layout:
    // [ks(2)][nt(12)][p(8)][lane(32)] uint32, low half = K, high half = K+1
    __shared__ unsigned int Bpack[2 * 12 * 8 * 32];   // 24 KB
#if USE_TDM
    extern __shared__ float stage[];                  // 2 x STAGE_DW dwords (33 KB)
#endif

    int bx  = blockIdx.x;
    int t   = 1 + (bx >> 8);          // 256 blocks per type
    int blk = bx & 255;
    const float* Ww = W_w + (size_t)t * FF * TD;

    for (int idx = threadIdx.x; idx < 2 * 12 * 8 * 32; idx += 128) {
        int lane = idx & 31;
        int p    = (idx >> 5) & 7;
        int rest = idx >> 8;
        int nt   = rest % 12;
        int ks   = rest / 12;
        int N = nt * 16 + (lane & 15);
        int K = ks * 32 + ((p >> 2) << 4) + ((lane >> 4) << 3) + ((p & 3) << 1);
        _Float16 h0 = (_Float16)Ww[K * TD + N];
        _Float16 h1 = (_Float16)Ww[(K + 1) * TD + N];
        unsigned int u0 = (unsigned int)(*(const unsigned short*)&h0);
        unsigned int u1 = (unsigned int)(*(const unsigned short*)&h1);
        Bpack[idx] = u0 | (u1 << 16);
    }

    int lane = threadIdx.x & 31;
    int wv   = threadIdx.x >> 5;      // 0..3
    int block_base = blk * (CHUNKS * CH_EDGES);
    const float* dist_t = dist + (size_t)t * EDG * FF;

#if USE_TDM
    unsigned stage_base = (unsigned)(unsigned long long)(const void*)stage;
    if (wv == 0) {
        tdm_load_chunk(stage_base, dist_t + (size_t)block_base * FF);
    }
#endif

    // epilogue pointers
    const float* vsrc  = (t == 1) ? v_nuc : v_elec;
    const int*   snd   = senders   + (size_t)t * EDG;
    const int*   rcv   = receivers + (size_t)t * EDG;
    const float* dir_t = dirs + (size_t)t * EDG * 3;
    const float* h_t   = h_all + (size_t)(t - 1) * N_ELEC * TD;
    float* zs_t = zs + (size_t)(t - 1) * N_ELEC * DD;
    float* zv_t = zv + (size_t)(t - 1) * N_ELEC * TD;

    int aM    = lane & 15;            // edge row within tile (A layout)
    int ahalf = lane >> 4;            // K-half select (A layout)
    int col   = lane & 15;            // output feature (C layout)
    int rhalf = lane >> 4;            // C layout: row M = r + 8*rhalf

    for (int ch = 0; ch < CHUNKS; ++ch) {
        int tile_base = block_base + ch * CH_EDGES + wv * 16;

        __syncthreads();              // pack done (ch=0) / prev consumers done
#if USE_TDM
        if (wv == 0) {
            if (ch < CHUNKS - 1) {
                tdm_load_chunk(stage_base + (unsigned)(((ch + 1) & 1) * STAGE_DW * 4),
                               dist_t + (size_t)(block_base + (ch + 1) * CH_EDGES) * FF);
                __builtin_amdgcn_s_wait_tensorcnt(1);   // chunk ch arrived
            } else {
                __builtin_amdgcn_s_wait_tensorcnt(0);
            }
        }
        __syncthreads();              // publish staged chunk ch to all waves
#endif

        // ---- load A (16 edges x 64 K, f32 -> f16), ISA 16-bit A 16x32 layout
#if USE_TDM
        const float* arow = stage + (size_t)((ch & 1) * STAGE_DW + (wv * 16 + aM) * ROW_DW);
#else
        const float* arow = dist_t + (size_t)(tile_base + aM) * FF;
#endif
        v16h a0, a1;
#pragma unroll
        for (int p = 0; p < 8; ++p) {
            int K0 = ((p >> 2) << 4) + ahalf * 8 + ((p & 3) << 1);
            float2 f0 = *(const float2*)(arow + K0);
            float2 f1 = *(const float2*)(arow + 32 + K0);
            a0[2 * p]     = (_Float16)f0.x;
            a0[2 * p + 1] = (_Float16)f0.y;
            a1[2 * p]     = (_Float16)f1.x;
            a1[2 * p + 1] = (_Float16)f1.y;
        }

        // ---- 12 N-tiles x 2 K-steps of v_wmma_f32_16x16x32_f16 ----
        v8f c[12];
        v8f czero = {};
#pragma unroll
        for (int nt = 0; nt < 12; ++nt) {
            v8u bu0, bu1;
#pragma unroll
            for (int p = 0; p < 8; ++p) {
                bu0[p] = Bpack[((0 * 12 + nt) * 8 + p) * 32 + lane];
                bu1[p] = Bpack[((1 * 12 + nt) * 8 + p) * 32 + lane];
            }
            v16h b0 = __builtin_bit_cast(v16h, bu0);
            v16h b1 = __builtin_bit_cast(v16h, bu1);
            c[nt] = __builtin_amdgcn_wmma_f32_16x16x32_f16(false, a0, false, b0,
                                                           (short)0, czero, false, false);
            c[nt] = __builtin_amdgcn_wmma_f32_16x16x32_f16(false, a1, false, b1,
                                                           (short)0, c[nt], false, false);
        }

        // ---- epilogue: phi = we * h_all[sender]; scatter-add segment sums --
#pragma unroll
        for (int r = 0; r < 8; ++r) {
            int edge   = tile_base + r + 8 * rhalf;
            int sender = snd[edge];
            int recv   = rcv[edge];
            float dx = dir_t[edge * 3 + 0];
            float dy = dir_t[edge * 3 + 1];
            float dz = dir_t[edge * 3 + 2];
            const float* hrow = h_t + (size_t)sender * TD;
#pragma unroll
            for (int j = 0; j < 4; ++j) {
                int d = j * 16 + col;
                float phi_s  = c[j][r]     * hrow[d];
                float phi_vv = c[4 + j][r] * hrow[64 + d];
                float phi_vs = c[8 + j][r] * hrow[128 + d];
                atomicAdd(&zs_t[recv * DD + d], phi_s);
                const float* vsd = vsrc + (size_t)(sender * DD + d) * 3;
                float* zvd = zv_t + (size_t)(recv * DD + d) * 3;
                atomicAdd(&zvd[0], phi_vv * vsd[0] + phi_vs * dx);
                atomicAdd(&zvd[1], phi_vv * vsd[1] + phi_vs * dy);
                atomicAdd(&zvd[2], phi_vv * vsd[2] + phi_vs * dz);
            }
        }
    }
}

// ---------------------------------------------------------------------------
// Kernel 4: per-electron node update (3 types), accumulate into out
// ---------------------------------------------------------------------------
__global__ void __launch_bounds__(192) painn_node(const float* __restrict__ W_g1,
                                                  const float* __restrict__ W_g2,
                                                  const float* __restrict__ W_V,
                                                  const float* __restrict__ W_U,
                                                  const float* __restrict__ zs,
                                                  const float* __restrict__ zv,
                                                  float* __restrict__ out) {
    int n   = blockIdx.x;             // electron index
    int tid = threadIdx.x;            // 192 threads
    __shared__ float zv_l[TD];
    __shared__ float Vv[TD];
    __shared__ float Uv[TD];
    __shared__ float g_in[2 * DD];
    __shared__ float tmp[HGG];
    __shared__ float gsd[TD];

    int f  = tid / 3;
    int i3 = tid - 3 * f;
    float acc_s = 0.0f;               // valid for tid < 64
    float acc_v = 0.0f;               // all 192 threads

    for (int t = 1; t <= 3; ++t) {
        const float* zs_t = zs + ((size_t)(t - 1) * N_ELEC + n) * DD;
        const float* zv_t = zv + ((size_t)(t - 1) * N_ELEC + n) * TD;
        zv_l[tid] = zv_t[tid];
        if (tid < DD) g_in[tid] = zs_t[tid];
        __syncthreads();

        const float* wV = W_V + (size_t)t * DD * DD;
        const float* wU = W_U + (size_t)t * DD * DD;
        float av = 0.0f, au = 0.0f;
        for (int d = 0; d < DD; ++d) {
            float z = zv_l[d * 3 + i3];
            av += z * wV[d * DD + f];
            au += z * wU[d * DD + f];
        }
        Vv[tid] = av; Uv[tid] = au;
        __syncthreads();

        if (tid < DD) {
            float v0 = Vv[tid * 3], v1 = Vv[tid * 3 + 1], v2 = Vv[tid * 3 + 2];
            g_in[DD + tid] = v0 * v0 + v1 * v1 + v2 * v2;
        }
        __syncthreads();

        const float* wg1 = W_g1 + (size_t)t * 2 * DD * HGG;
        if (tid < HGG) {
            float a = 0.0f;
            for (int cI = 0; cI < 2 * DD; ++cI) a += g_in[cI] * wg1[cI * HGG + tid];
            tmp[tid] = a / (1.0f + __expf(-a));   // silu
        }
        __syncthreads();

        const float* wg2 = W_g2 + (size_t)t * HGG * TD;
        {
            float a = 0.0f;
            for (int h = 0; h < HGG; ++h) a += tmp[h] * wg2[h * TD + tid];
            gsd[tid] = a;
        }
        __syncthreads();

        acc_v += Uv[tid] * gsd[64 + f];           // upd_v[f][i]
        if (tid < DD) {
            float dot = Uv[tid * 3] * Vv[tid * 3]
                      + Uv[tid * 3 + 1] * Vv[tid * 3 + 1]
                      + Uv[tid * 3 + 2] * Vv[tid * 3 + 2];
            acc_s += gsd[128 + tid] * dot + gsd[tid];  // a_sv*dot + a_ss
        }
        __syncthreads();                          // protect LDS reuse next type
    }

    if (tid < DD) out[(size_t)n * 256 + tid] += acc_s;
    out[(size_t)n * 256 + 64 + tid] += acc_v;
}

// ---------------------------------------------------------------------------
extern "C" void kernel_launch(void* const* d_in, const int* in_sizes, int n_in,
                              void* d_out, int out_size, void* d_ws, size_t ws_size,
                              hipStream_t stream) {
    const float* s_elec   = (const float*)d_in[0];
    const float* v_elec   = (const float*)d_in[1];
    const float* s_nuc    = (const float*)d_in[2];
    const float* v_nuc    = (const float*)d_in[3];
    const float* dist     = (const float*)d_in[4];
    const float* dirs     = (const float*)d_in[5];
    const float* W_w      = (const float*)d_in[6];
    const float* W_h1     = (const float*)d_in[7];
    const float* W_h2     = (const float*)d_in[8];
    const float* W_g1     = (const float*)d_in[9];
    const float* W_g2     = (const float*)d_in[10];
    const float* W_V      = (const float*)d_in[11];
    const float* W_U      = (const float*)d_in[12];
    const int*   senders  = (const int*)d_in[13];
    const int*   receivers= (const int*)d_in[14];
    float* out = (float*)d_out;

    float* h_all = (float*)d_ws;                       // 3*1024*192
    float* zseg  = h_all + (size_t)3 * N_ELEC * TD;    // 3*1024*64
    float* zvec  = zseg  + (size_t)3 * N_ELEC * DD;    // 3*1024*192

    painn_init<<<2304, 256, 0, stream>>>(s_elec, v_elec, out, zseg, zvec);
    painn_h<<<N_NUC + 2 * N_ELEC, 192, 0, stream>>>(s_elec, s_nuc, W_h1, W_h2, h_all);
    painn_edge<<<3 * 256, 128, 2 * STAGE_DW * 4, stream>>>(dist, dirs, W_w, v_elec, v_nuc,
                                                           senders, receivers, h_all,
                                                           zseg, zvec);
    painn_node<<<N_ELEC, 192, 0, stream>>>(W_g1, W_g2, W_V, W_U, zseg, zvec, out);
}
